// GRU_11149735100891
// MI455X (gfx1250) — compile-verified, matrix operand order checked by reading, not run on hardware
//
#include <hip/hip_runtime.h>
#include <hip/hip_bf16.h>
#include <cstdint>

typedef __attribute__((ext_vector_type(16))) __bf16 v16bf;
typedef __attribute__((ext_vector_type(8)))  __bf16 v8bf;
typedef __attribute__((ext_vector_type(8)))  float  v8f;
typedef __attribute__((__vector_size__(4 * sizeof(int)))) int v4i;

#define BDIM 8192
#define DDIM 1024
#define HDIM 1024
#define BM 128
#define BN 64
#define BK 32
#define LDA_S 40   // padded bf16 row stride for A tile (32 + 8)
#define LDB_S 40   // padded bf16 row stride for (pre-transposed) B tiles

// ---------------- fp32 -> bf16 (RNE) ----------------
static __device__ __forceinline__ uint16_t bf16_bits(float f) {
  uint32_t u = __float_as_uint(f);
  u += 0x7FFFu + ((u >> 16) & 1u);
  return (uint16_t)(u >> 16);
}
static __device__ __forceinline__ __bf16 cvt_bf16(float f) {
  uint16_t s = bf16_bits(f);
  __bf16 b;
  __builtin_memcpy(&b, &s, sizeof(b));
  return b;
}

// ---------------- async global->LDS copy (16B per lane) ----------------
#if defined(__gfx1250__) && __has_builtin(__builtin_amdgcn_global_load_async_to_lds_b128)
#define HAVE_ASYNC_LDS 1
static __device__ __forceinline__ void cp16_g2l(const __bf16* g, __bf16* l) {
  __builtin_amdgcn_global_load_async_to_lds_b128(
      (__attribute__((address_space(1))) v4i*)g,
      (__attribute__((address_space(3))) v4i*)l,
      0, 0);
}
#else
#define HAVE_ASYNC_LDS 0
static __device__ __forceinline__ void cp16_g2l(const __bf16* g, __bf16* l) {
  *(v8bf*)l = *(const v8bf*)g;   // global_load_b128 + ds_store_b128
}
#endif

#if defined(__gfx1250__) && __has_builtin(__builtin_amdgcn_s_wait_asynccnt)
#define WAIT_ASYNC(n) __builtin_amdgcn_s_wait_asynccnt(n)
#elif defined(__gfx1250__) && HAVE_ASYNC_LDS
#define WAIT_ASYNC(n) asm volatile("s_wait_asynccnt %0" ::"n"(n) : "memory")
#else
#define WAIT_ASYNC(n) ((void)0)
#endif

// ---------------- fragment gather from LDS ----------------
static __device__ __forceinline__ v16bf load_frag(const __bf16* p, int half) {
  v8bf lo = *(const v8bf*)(p + half * 8);
  v8bf hi = *(const v8bf*)(p + 16 + half * 8);
  return __builtin_shufflevector(lo, hi,
                                 0, 1, 2, 3, 4, 5, 6, 7,
                                 8, 9, 10, 11, 12, 13, 14, 15);
}

static __device__ __forceinline__ float sigmoidf_fast(float x) {
  return 1.0f / (1.0f + __expf(-x));
}
static __device__ __forceinline__ float tanhf_fast(float x) {
  float e = __expf(-2.0f * x);
  return (1.0f - e) / (1.0f + e);
}

// =====================================================================
// Pre-pass 1: pack fp32 -> bf16 (flat copy), 4 elements / thread
// =====================================================================
__global__ __launch_bounds__(256)
void pack_bf16_kernel(const float* __restrict__ src, uint16_t* __restrict__ dst) {
  size_t idx = (size_t)blockIdx.x * 256 + threadIdx.x;
  const float4 v = ((const float4*)src)[idx];
  ushort4 o;
  o.x = bf16_bits(v.x); o.y = bf16_bits(v.y);
  o.z = bf16_bits(v.z); o.w = bf16_bits(v.w);
  ((ushort4*)dst)[idx] = o;
}

// =====================================================================
// Pre-pass 2: transpose-convert the six 1024x1024 weights into
// Wt[slot][n][k] bf16, slot order: wz, wr, wx, uz, ur, wh
// =====================================================================
__global__ __launch_bounds__(256)
void wtrans_kernel(const float* __restrict__ wz, const float* __restrict__ wr,
                   const float* __restrict__ wx, const float* __restrict__ uz,
                   const float* __restrict__ ur, const float* __restrict__ wh,
                   uint16_t* __restrict__ Wt) {
  __shared__ float tile[32][33];
  const float* src;
  switch (blockIdx.z) {
    case 0: src = wz; break;
    case 1: src = wr; break;
    case 2: src = wx; break;
    case 3: src = uz; break;
    case 4: src = ur; break;
    default: src = wh; break;
  }
  uint16_t* dst = Wt + (size_t)blockIdx.z * (1024u * 1024u);
  const int n0 = blockIdx.x * 32;
  const int k0 = blockIdx.y * 32;
  const int tx = threadIdx.x;   // 0..31
  const int ty = threadIdx.y;   // 0..7
#pragma unroll
  for (int j = 0; j < 4; ++j) {
    int r = ty + j * 8;
    tile[r][tx] = src[(size_t)(k0 + r) * HDIM + n0 + tx];
  }
  __syncthreads();
#pragma unroll
  for (int j = 0; j < 4; ++j) {
    int r = ty + j * 8;
    dst[(size_t)(n0 + r) * 1024 + k0 + tx] = bf16_bits(tile[tx][r]);
  }
}

// =====================================================================
// Main fused GRU GEMM: bf16 WMMA, async triple-buffered LDS staging
// with prefetch distance 2 and one barrier per K-step.
// Abf: [2][8192][1024] bf16 (inp then state)
// Wt : [6][1024][1024] bf16, transposed ([n][k]); slots {wz,wr,wx,uz,ur,wh}
// =====================================================================
__global__ __launch_bounds__(256, 1)
void gru_wmma_async(const uint16_t* __restrict__ Abf_, const uint16_t* __restrict__ Wt_,
                    const float* __restrict__ state,
                    const float* __restrict__ bx, const float* __restrict__ br,
                    const float* __restrict__ bz, float* __restrict__ out)
{
  const __bf16* Abf = (const __bf16*)Abf_;
  const __bf16* Wt  = (const __bf16*)Wt_;

  __shared__ __bf16 sA[3][BM * LDA_S];       // 3 x 10 KB
  __shared__ __bf16 sB[3][3][BN * LDB_S];    // 3 x 15 KB

  const int tid  = threadIdx.x;
  const int lane = tid & 31;
  const int wave = tid >> 5;
  const int half = lane >> 4;
  const int l16  = lane & 15;
  const int wm   = wave >> 1;   // 0..3 -> M offset wm*32
  const int wn   = wave & 1;    // 0..1 -> N offset wn*32

  const int m0 = blockIdx.x * BM;
  const int n0 = blockIdx.y * BN;

  v8f accZ[2][2], accR[2][2], accXH[2][2], accHH[2][2];
#pragma unroll
  for (int ms = 0; ms < 2; ++ms)
#pragma unroll
    for (int ns = 0; ns < 2; ++ns) {
      accZ[ms][ns]  = (v8f){0.f, 0.f, 0.f, 0.f, 0.f, 0.f, 0.f, 0.f};
      accR[ms][ns]  = accZ[ms][ns];
      accXH[ms][ns] = accZ[ms][ns];
      accHH[ms][ns] = accZ[ms][ns];
    }

  // Per-thread staging coordinates (5 x 16B chunks per tile-step)
  const int arow0 = tid >> 2;                 // A chunk 0: rows 0..63
  const int arow1 = (tid + 256) >> 2;         // A chunk 1: rows 64..127
  const int ac    = (tid & 3) * 8;            // k-chunk within row
  const int brow  = tid >> 2;                 // B: 64 rows x 4 chunks

  const int aoff0 = arow0 * LDA_S + ac;       // LDS offsets (constant per thread)
  const int aoff1 = arow1 * LDA_S + ac;
  const int boff  = brow * LDB_S + ac;

  // Incrementally-advanced global pointers (advance 32 elems per staged tile,
  // one phase jump after tile 31).
  const __bf16* pA0 = Abf + (size_t)(m0 + arow0) * DDIM + ac;
  const __bf16* pA1 = Abf + (size_t)(m0 + arow1) * DDIM + ac;
  const __bf16* pB  = Wt + (size_t)(n0 + brow) * 1024 + ac;   // gate 0, phase 0

  auto stage = [&](int buf, int tile) {
    cp16_g2l(pA0, &sA[buf][aoff0]);
    cp16_g2l(pA1, &sA[buf][aoff1]);
#pragma unroll
    for (int g = 0; g < 3; ++g)
      cp16_g2l(pB + (size_t)g * (1024u * 1024u), &sB[buf][g][boff]);
    if (tile == 31) {                 // phase seam: inp->state, slots 0..2 -> 3..5
      pA0 += (size_t)BDIM * DDIM - 992;
      pA1 += (size_t)BDIM * DDIM - 992;
      pB  += (size_t)3 * 1024 * 1024 - 992;
    } else {
      pA0 += BK; pA1 += BK; pB += BK;
    }
  };

  stage(0, 0);
  stage(1, 1);

  int cb = 0;        // compute buffer = t % 3
  int sb = 2;        // stage buffer   = (t+2) % 3
  for (int t = 0; t < 64; ++t) {
    if (t < 62) {
      WAIT_ASYNC(5);          // own tile-t copies done (t+1 may be in flight)
      __syncthreads();        // all waves' tile-t copies visible
      stage(sb, t + 2);       // overwrites buffer last read at t-1 (pre-barrier)
    } else {
      WAIT_ASYNC(0);
      __syncthreads();
    }

    const __bf16* cA = &sA[cb][0];
    const __bf16* cB = &sB[cb][0][0];

    v16bf a[2], fz[2], fr[2], fh[2];
#pragma unroll
    for (int ms = 0; ms < 2; ++ms)
      a[ms] = load_frag(cA + (wm * 32 + ms * 16 + l16) * LDA_S, half);
#pragma unroll
    for (int ns = 0; ns < 2; ++ns) {
      int col = wn * 32 + ns * 16 + l16;
      fz[ns] = load_frag(cB + 0 * (BN * LDB_S) + col * LDB_S, half);
      fr[ns] = load_frag(cB + 1 * (BN * LDB_S) + col * LDB_S, half);
      fh[ns] = load_frag(cB + 2 * (BN * LDB_S) + col * LDB_S, half);
    }

    const bool firstHalf = (t < 32);
#pragma unroll
    for (int ms = 0; ms < 2; ++ms)
#pragma unroll
      for (int ns = 0; ns < 2; ++ns) {
        accZ[ms][ns] = __builtin_amdgcn_wmma_f32_16x16x32_bf16(
            false, a[ms], false, fz[ns], (short)0, accZ[ms][ns], false, false);
        accR[ms][ns] = __builtin_amdgcn_wmma_f32_16x16x32_bf16(
            false, a[ms], false, fr[ns], (short)0, accR[ms][ns], false, false);
        if (firstHalf)
          accXH[ms][ns] = __builtin_amdgcn_wmma_f32_16x16x32_bf16(
              false, a[ms], false, fh[ns], (short)0, accXH[ms][ns], false, false);
        else
          accHH[ms][ns] = __builtin_amdgcn_wmma_f32_16x16x32_bf16(
              false, a[ms], false, fh[ns], (short)0, accHH[ms][ns], false, false);
      }

    cb = (cb == 2) ? 0 : cb + 1;
    sb = (sb == 2) ? 0 : sb + 1;
  }

  // ---- fused GRU epilogue ----
#pragma unroll
  for (int ms = 0; ms < 2; ++ms) {
#pragma unroll
    for (int ns = 0; ns < 2; ++ns) {
      const int nn = n0 + wn * 32 + ns * 16 + l16;
      const int mb = m0 + wm * 32 + ms * 16 + half * 8;
      const float bzv = bz[nn];
      const float brv = br[nn];
      const float bxv = bx[nn];
#pragma unroll
      for (int i = 0; i < 8; ++i) {
        const int m = mb + i;
        float z  = sigmoidf_fast(accZ[ms][ns][i] + bzv);
        float r  = sigmoidf_fast(accR[ms][ns][i] + brv);
        float hc = tanhf_fast(accXH[ms][ns][i] + bxv + accHH[ms][ns][i] * r);
        float st = state[(size_t)m * HDIM + nn];
        out[(size_t)m * HDIM + nn] = (1.0f - z) * hc + st * z;
      }
    }
  }
}

// =====================================================================
// Fallback (round-1 kernel): in-loop conversion, no workspace needed.
// =====================================================================
__global__ __launch_bounds__(256, 1)
void gru_fused_wmma(const float* __restrict__ inp, const float* __restrict__ state,
                    const float* __restrict__ wz,  const float* __restrict__ uz,
                    const float* __restrict__ wr,  const float* __restrict__ ur,
                    const float* __restrict__ wx,  const float* __restrict__ wh,
                    const float* __restrict__ bx,  const float* __restrict__ br,
                    const float* __restrict__ bz,  float* __restrict__ out)
{
  __shared__ __bf16 sA[BM * LDA_S];
  __shared__ __bf16 sB[3][BN * LDB_S];

  const int tid  = threadIdx.x;
  const int lane = tid & 31;
  const int wave = tid >> 5;
  const int half = lane >> 4;
  const int l16  = lane & 15;
  const int wm   = wave >> 1;
  const int wn   = wave & 1;

  const int m0 = blockIdx.x * BM;
  const int n0 = blockIdx.y * BN;

  v8f accZ[2][2], accR[2][2], accXH[2][2], accHH[2][2];
#pragma unroll
  for (int ms = 0; ms < 2; ++ms)
#pragma unroll
    for (int ns = 0; ns < 2; ++ns) {
      accZ[ms][ns]  = (v8f){0.f, 0.f, 0.f, 0.f, 0.f, 0.f, 0.f, 0.f};
      accR[ms][ns]  = accZ[ms][ns];
      accXH[ms][ns] = accZ[ms][ns];
      accHH[ms][ns] = accZ[ms][ns];
    }

#pragma unroll
  for (int ph = 0; ph < 2; ++ph) {
    const float* Ag  = ph ? state : inp;
    const float* Bg0 = ph ? uz : wz;
    const float* Bg1 = ph ? ur : wr;
    const float* Bg2 = ph ? wh : wx;

    for (int kk = 0; kk < DDIM; kk += BK) {
      __syncthreads();
#pragma unroll
      for (int i = 0; i < 4; ++i) {
        int idx = tid + i * 256;
        int row = idx >> 3;
        int c4  = (idx & 7) * 4;
        const float4 v = *(const float4*)(Ag + (size_t)(m0 + row) * DDIM + kk + c4);
        __bf16* d = &sA[row * LDA_S + c4];
        d[0] = cvt_bf16(v.x); d[1] = cvt_bf16(v.y);
        d[2] = cvt_bf16(v.z); d[3] = cvt_bf16(v.w);
      }
      const float* Bg[3] = {Bg0, Bg1, Bg2};
#pragma unroll
      for (int g = 0; g < 3; ++g) {
#pragma unroll
        for (int i = 0; i < 2; ++i) {
          int idx = tid + i * 256;
          int k   = idx >> 4;
          int n4  = (idx & 15) * 4;
          const float4 v = *(const float4*)(Bg[g] + (size_t)(kk + k) * HDIM + n0 + n4);
          sB[g][(n4 + 0) * LDB_S + k] = cvt_bf16(v.x);
          sB[g][(n4 + 1) * LDB_S + k] = cvt_bf16(v.y);
          sB[g][(n4 + 2) * LDB_S + k] = cvt_bf16(v.z);
          sB[g][(n4 + 3) * LDB_S + k] = cvt_bf16(v.w);
        }
      }
      __syncthreads();

      v16bf a[2], fz[2], fr[2], fh[2];
#pragma unroll
      for (int ms = 0; ms < 2; ++ms)
        a[ms] = load_frag(&sA[(wm * 32 + ms * 16 + l16) * LDA_S], half);
#pragma unroll
      for (int ns = 0; ns < 2; ++ns) {
        int col = wn * 32 + ns * 16 + l16;
        fz[ns] = load_frag(&sB[0][col * LDB_S], half);
        fr[ns] = load_frag(&sB[1][col * LDB_S], half);
        fh[ns] = load_frag(&sB[2][col * LDB_S], half);
      }
#pragma unroll
      for (int ms = 0; ms < 2; ++ms)
#pragma unroll
        for (int ns = 0; ns < 2; ++ns) {
          accZ[ms][ns] = __builtin_amdgcn_wmma_f32_16x16x32_bf16(
              false, a[ms], false, fz[ns], (short)0, accZ[ms][ns], false, false);
          accR[ms][ns] = __builtin_amdgcn_wmma_f32_16x16x32_bf16(
              false, a[ms], false, fr[ns], (short)0, accR[ms][ns], false, false);
          if (ph == 0)
            accXH[ms][ns] = __builtin_amdgcn_wmma_f32_16x16x32_bf16(
                false, a[ms], false, fh[ns], (short)0, accXH[ms][ns], false, false);
          else
            accHH[ms][ns] = __builtin_amdgcn_wmma_f32_16x16x32_bf16(
                false, a[ms], false, fh[ns], (short)0, accHH[ms][ns], false, false);
        }
    }
  }

#pragma unroll
  for (int ms = 0; ms < 2; ++ms) {
#pragma unroll
    for (int ns = 0; ns < 2; ++ns) {
      const int nn = n0 + wn * 32 + ns * 16 + l16;
      const int mb = m0 + wm * 32 + ms * 16 + half * 8;
      const float bzv = bz[nn];
      const float brv = br[nn];
      const float bxv = bx[nn];
#pragma unroll
      for (int i = 0; i < 8; ++i) {
        const int m = mb + i;
        float z  = sigmoidf_fast(accZ[ms][ns][i] + bzv);
        float r  = sigmoidf_fast(accR[ms][ns][i] + brv);
        float hc = tanhf_fast(accXH[ms][ns][i] + bxv + accHH[ms][ns][i] * r);
        float st = state[(size_t)m * HDIM + nn];
        out[(size_t)m * HDIM + nn] = (1.0f - z) * hc + st * z;
      }
    }
  }
}

extern "C" void kernel_launch(void* const* d_in, const int* in_sizes, int n_in,
                              void* d_out, int out_size, void* d_ws, size_t ws_size,
                              hipStream_t stream) {
  // setup_inputs order: inp, state, wx, bx, wh, wr, ur, uz, wz, br, bz
  const float* inp   = (const float*)d_in[0];
  const float* state = (const float*)d_in[1];
  const float* wx    = (const float*)d_in[2];
  const float* bx    = (const float*)d_in[3];
  const float* wh    = (const float*)d_in[4];
  const float* wr    = (const float*)d_in[5];
  const float* ur    = (const float*)d_in[6];
  const float* uz    = (const float*)d_in[7];
  const float* wz    = (const float*)d_in[8];
  const float* br    = (const float*)d_in[9];
  const float* bz    = (const float*)d_in[10];
  float* out = (float*)d_out;

  const size_t aBytes = (size_t)2 * BDIM * DDIM * sizeof(uint16_t);   // 32 MB
  const size_t wBytes = (size_t)6 * 1024 * 1024 * sizeof(uint16_t);   // 12 MB

  if (ws_size >= aBytes + wBytes) {
    uint16_t* Abf = (uint16_t*)d_ws;
    uint16_t* Wt  = (uint16_t*)((char*)d_ws + aBytes);

    // pre-pass: pack activations, transpose+pack weights (all L2-resident)
    const int packBlocks = (BDIM * DDIM / 4) / 256;   // 8192
    pack_bf16_kernel<<<packBlocks, 256, 0, stream>>>(inp, Abf);
    pack_bf16_kernel<<<packBlocks, 256, 0, stream>>>(state, Abf + (size_t)BDIM * DDIM);
    wtrans_kernel<<<dim3(32, 32, 6), dim3(32, 8), 0, stream>>>(wz, wr, wx, uz, ur, wh, Wt);

    dim3 grid(BDIM / BM, HDIM / BN);
    gru_wmma_async<<<grid, dim3(256), 0, stream>>>(Abf, Wt, state, bx, br, bz, out);
  } else {
    dim3 grid(BDIM / BM, HDIM / BN);
    gru_fused_wmma<<<grid, dim3(256), 0, stream>>>(inp, state, wz, uz, wr, ur,
                                                   wx, wh, bx, br, bz, out);
  }
}
